// SCA_attention_72859825209774
// MI455X (gfx1250) — compile-verified, matrix operand order checked by reading, not run on hardware
//
#include <hip/hip_runtime.h>

// ---------------------------------------------------------------------------
// CDNA5 WMMA types (wave32): v_wmma_f32_16x16x32_bf16
// ---------------------------------------------------------------------------
typedef __attribute__((ext_vector_type(16))) __bf16 v16bf;
typedef __attribute__((ext_vector_type(8)))  __bf16 v8bf;
typedef __attribute__((ext_vector_type(4)))  __bf16 v4bf;
typedef __attribute__((ext_vector_type(8)))  float  v8f;

// ---------------------------------------------------------------------------
// Batched GEMM: C[b] = op(A[b]) * B[b] (+ bias), fp32 in/out, bf16 WMMA math.
//   TA==0: A is [M][lda] row-major.
//   TA==1: A stored [K][lda]; logical A[m][k] = A[k*lda + m].
//   B always [K][ldb] row-major.
// 256 threads = 8 waves (2 M x 4 N); wave tile 32x32 (2x2 WMMA frags).
// Block tile BM=64, BN=128, BK=32; double-buffered LDS, 1 barrier / K-step.
// All problem dims are exact multiples of the tiles.
// ---------------------------------------------------------------------------
template <int TA>
__global__ __launch_bounds__(256)
void gemm_wmma(const float* __restrict__ A, const float* __restrict__ B,
               const float* __restrict__ bias, float* __restrict__ C,
               int K, int lda, int ldb, int ldc,
               long long sA, long long sB, long long sC)
{
  __shared__ __align__(16) __bf16 As[2][64][40];    // [buf][m][k], 80B rows
  __shared__ __align__(16) __bf16 Bs[2][128][40];   // [buf][n][k]

  const float* __restrict__ Ab = A + sA * (long long)blockIdx.z;
  const float* __restrict__ Bb = B + sB * (long long)blockIdx.z;
  float* __restrict__       Cb = C + sC * (long long)blockIdx.z;

  const int tid  = threadIdx.x;
  const int lane = tid & 31;
  const int w    = tid >> 5;
  const int wm   = w >> 2;          // 0..1 (M)
  const int wn   = w & 3;           // 0..3 (N)
  const int row0 = blockIdx.y * 64;
  const int col0 = blockIdx.x * 128;

  // ---- per-thread staging coordinates (int32 running offsets) ----
  // A: TA==0 -> thread owns (row a_i, 4 consecutive k) via one b128 load.
  //    TA==1 -> thread owns (row a_i, 4 consecutive k) via 4 strided b32 loads
  //             (coalesced along m).  Both store one contiguous ds_store_b64.
  int a_i[2], a_k[2], a_off[2];
#pragma unroll
  for (int t = 0; t < 2; ++t) {
    const int idx = tid + t * 256;            // 512 groups of 4-along-k
    if (TA == 0) { a_i[t] = idx >> 3;  a_k[t] = (idx & 7) << 2;
                   a_off[t] = (row0 + a_i[t]) * lda + a_k[t]; }
    else         { a_i[t] = idx & 63;  a_k[t] = (idx >> 6) << 2;
                   a_off[t] = a_k[t] * lda + row0 + a_i[t]; }
  }
  // B: thread owns (col b_n, 4 consecutive k): 4 coalesced b32 loads
  //    (consecutive threads -> consecutive n) + one ds_store_b64.
  int b_n[4], b_k[4], b_off[4];
#pragma unroll
  for (int t = 0; t < 4; ++t) {
    const int idx = tid + t * 256;            // 1024 groups of 4-along-k
    b_n[t] = idx & 127;  b_k[t] = (idx >> 7) << 2;
    b_off[t] = b_k[t] * ldb + col0 + b_n[t];
  }
  const int a_step = (TA == 0) ? 32 : 32 * lda;
  const int b_step = 32 * ldb;

  float ra[2][4], rb[4][4];

  auto LOAD = [&]() {   // issue global loads for the next tile into registers
    if (TA == 0) {
#pragma unroll
      for (int t = 0; t < 2; ++t) {
        const float4 v = *reinterpret_cast<const float4*>(&Ab[a_off[t]]);
        ra[t][0] = v.x; ra[t][1] = v.y; ra[t][2] = v.z; ra[t][3] = v.w;
        a_off[t] += a_step;
      }
    } else {
#pragma unroll
      for (int t = 0; t < 2; ++t) {
#pragma unroll
        for (int j = 0; j < 4; ++j) ra[t][j] = Ab[a_off[t] + j * lda];
        a_off[t] += a_step;
      }
    }
#pragma unroll
    for (int t = 0; t < 4; ++t) {
#pragma unroll
      for (int j = 0; j < 4; ++j) rb[t][j] = Bb[b_off[t] + j * ldb];
      b_off[t] += b_step;
    }
  };

  auto STORE = [&](int buf) {   // convert + contiguous ds_store_b64
#pragma unroll
    for (int t = 0; t < 2; ++t) {
      v4bf h;
      h[0] = (__bf16)ra[t][0]; h[1] = (__bf16)ra[t][1];
      h[2] = (__bf16)ra[t][2]; h[3] = (__bf16)ra[t][3];
      *reinterpret_cast<v4bf*>(&As[buf][a_i[t]][a_k[t]]) = h;
    }
#pragma unroll
    for (int t = 0; t < 4; ++t) {
      v4bf h;
      h[0] = (__bf16)rb[t][0]; h[1] = (__bf16)rb[t][1];
      h[2] = (__bf16)rb[t][2]; h[3] = (__bf16)rb[t][3];
      *reinterpret_cast<v4bf*>(&Bs[buf][b_n[t]][b_k[t]]) = h;
    }
  };

  // ISA 16-bit A 16x32 layout: row = lane&15; lanes 0-15 hold K 0-7/16-23,
  // lanes 16-31 hold K 8-15/24-31.  B mirrored on N.
  const int fr = lane & 15;
  const int kb = (lane >> 4) * 8;   // K-half base
  const int mb = kb;                // C/D row base

  v8f acc[2][2] = {};

  LOAD();
  STORE(0);
  int p = 0;
  const int nk = K >> 5;
  for (int kt = 0; kt < nk; ++kt) {
    __syncthreads();                          // buf p ready for all waves
    const bool more = (kt + 1 < nk);
    if (more) {
      LOAD();                                 // global loads overlap the WMMAs
      __builtin_prefetch(&Bb[b_off[0]], 0, 0);  // tile kt+2 -> global_prefetch_b8
    }

    // ---- fragments from buf p: two contiguous ds_load_b128 each ----
    v16bf af[2], bf[2];
#pragma unroll
    for (int f = 0; f < 2; ++f) {
      const int r = wm * 32 + f * 16 + fr;
      const v8bf alo = *reinterpret_cast<const v8bf*>(&As[p][r][kb]);
      const v8bf ahi = *reinterpret_cast<const v8bf*>(&As[p][r][16 + kb]);
      const int c = wn * 32 + f * 16 + fr;
      const v8bf blo = *reinterpret_cast<const v8bf*>(&Bs[p][c][kb]);
      const v8bf bhi = *reinterpret_cast<const v8bf*>(&Bs[p][c][16 + kb]);
#pragma unroll
      for (int j = 0; j < 8; ++j) {
        af[f][j] = alo[j]; af[f][j + 8] = ahi[j];
        bf[f][j] = blo[j]; bf[f][j + 8] = bhi[j];
      }
    }
#pragma unroll
    for (int fi = 0; fi < 2; ++fi)
#pragma unroll
      for (int fj = 0; fj < 2; ++fj)
        acc[fi][fj] = __builtin_amdgcn_wmma_f32_16x16x32_bf16(
            false, af[fi], false, bf[fj], (short)0, acc[fi][fj], false, false);

    if (more) STORE(p ^ 1);                   // waits for loads *after* WMMAs
    p ^= 1;
  }

  // ---- store C (D layout: VGPR r -> row mb+r, col lane&15) ----
#pragma unroll
  for (int fi = 0; fi < 2; ++fi) {
#pragma unroll
    for (int fj = 0; fj < 2; ++fj) {
      const int gcol = col0 + wn * 32 + fj * 16 + (lane & 15);
#pragma unroll
      for (int r = 0; r < 8; ++r) {
        const int grow = row0 + wm * 32 + fi * 16 + mb + r;
        float v = acc[fi][fj][r];
        if (bias) v += bias[grow];
        Cb[grow * ldc + gcol] = v;
      }
    }
  }
}

// ---------------------------------------------------------------------------
// Column softmax over K=192 of attT [16][192][4096]; one thread per (b, n).
// ---------------------------------------------------------------------------
__global__ __launch_bounds__(256)
void softmax_k(float* __restrict__ att)
{
  const int col = blockIdx.x * 256 + threadIdx.x;       // 0 .. 16*4096-1
  const int b = col >> 12, n = col & 4095;
  float* p = att + ((long long)b * 192) * 4096 + n;
  float m = -1e30f;
  for (int k = 0; k < 192; ++k) m = fmaxf(m, p[k * 4096]);
  float s = 0.f;
  for (int k = 0; k < 192; ++k) s += __expf(p[k * 4096] - m);
  const float inv = 1.f / s;
  for (int k = 0; k < 192; ++k) p[k * 4096] = __expf(p[k * 4096] - m) * inv;
}

// Row sums over n (contiguous): rs[b*192+k] = sum_n attT[b][k][n]
__global__ __launch_bounds__(256)
void rowsum_kernel(const float* __restrict__ att, float* __restrict__ rs)
{
  const float4* p = reinterpret_cast<const float4*>(att + (long long)blockIdx.x * 4096);
  float s = 0.f;
  for (int i = threadIdx.x; i < 1024; i += 256) {
    const float4 v = p[i];
    s += v.x + v.y + v.z + v.w;
  }
#pragma unroll
  for (int o = 16; o > 0; o >>= 1) s += __shfl_xor(s, o);
  __shared__ float red[8];
  if ((threadIdx.x & 31) == 0) red[threadIdx.x >> 5] = s;
  __syncthreads();
  if (threadIdx.x == 0) {
    float t = 0.f;
#pragma unroll
    for (int i = 0; i < 8; ++i) t += red[i];
    rs[blockIdx.x] = t;
  }
}

// attT[b][k][:] *= 1/(1e-6 + rs[b*192+k])   (float4; row id = i>>10)
__global__ void scale_att(float4* __restrict__ att, const float* __restrict__ rs,
                          long long total4)
{
  long long i = (long long)blockIdx.x * blockDim.x + threadIdx.x;
  const long long stride = (long long)gridDim.x * blockDim.x;
  for (; i < total4; i += stride) {
    const float d = 1.f / (1e-6f + rs[(int)(i >> 10)]);
    float4 v = att[i];
    v.x *= d; v.y *= d; v.z *= d; v.w *= d;
    att[i] = v;
  }
}

// ---------------------------------------------------------------------------
// BN batch stats over (B=16, N=4096) per channel of [16][512][4096].
// stats[2c]=mean, stats[2c+1]=rsqrt(var+eps)  (biased var, matches jnp)
// ---------------------------------------------------------------------------
__global__ __launch_bounds__(256)
void bn_stats(const float* __restrict__ y, float* __restrict__ stats)
{
  const int c = blockIdx.x;
  const float4* y4 = reinterpret_cast<const float4*>(y);
  float s = 0.f, q = 0.f;
  for (int i = threadIdx.x; i < 16 * 1024; i += 256) {
    const int b = i >> 10, n4 = i & 1023;
    const float4 v = y4[(b * 512 + c) * 1024 + n4];
    s += v.x + v.y + v.z + v.w;
    q += v.x * v.x + v.y * v.y + v.z * v.z + v.w * v.w;
  }
#pragma unroll
  for (int o = 16; o > 0; o >>= 1) { s += __shfl_xor(s, o); q += __shfl_xor(q, o); }
  __shared__ float rs[8], rq[8];
  if ((threadIdx.x & 31) == 0) { rs[threadIdx.x >> 5] = s; rq[threadIdx.x >> 5] = q; }
  __syncthreads();
  if (threadIdx.x == 0) {
    float S = 0.f, Q = 0.f;
#pragma unroll
    for (int i = 0; i < 8; ++i) { S += rs[i]; Q += rq[i]; }
    const float inv = 1.f / (16.f * 4096.f);
    const float mean = S * inv;
    const float var = Q * inv - mean * mean;
    stats[2 * c]     = mean;
    stats[2 * c + 1] = rsqrtf(var + 1e-5f);
  }
}

// out = relu(g*(y-mean)*invstd + beta) [+ res]   float4 over [16*512*1024] f4
__global__ void bn_relu_res(const float4* __restrict__ y, const float* __restrict__ stats,
                            const float* __restrict__ g, const float* __restrict__ be,
                            const float4* __restrict__ res, float4* __restrict__ out,
                            long long total4)
{
  long long i = (long long)blockIdx.x * blockDim.x + threadIdx.x;
  const long long stride = (long long)gridDim.x * blockDim.x;
  for (; i < total4; i += stride) {
    const int c = (int)(i >> 10) & 511;                   // 1024 f4 per channel row
    const float mean = stats[2 * c], inv = stats[2 * c + 1];
    const float ga = g[c], bb = be[c];
    float4 v = y[i];
    v.x = fmaxf(ga * (v.x - mean) * inv + bb, 0.f);
    v.y = fmaxf(ga * (v.y - mean) * inv + bb, 0.f);
    v.z = fmaxf(ga * (v.z - mean) * inv + bb, 0.f);
    v.w = fmaxf(ga * (v.w - mean) * inv + bb, 0.f);
    if (res) {
      const float4 r = res[i];
      v.x += r.x; v.y += r.y; v.z += r.z; v.w += r.w;
    }
    out[i] = v;
  }
}

// ---------------------------------------------------------------------------
// Orchestration
// ---------------------------------------------------------------------------
extern "C" void kernel_launch(void* const* d_in, const int* in_sizes, int n_in,
                              void* d_out, int out_size, void* d_ws, size_t ws_size,
                              hipStream_t stream)
{
  (void)in_sizes; (void)n_in; (void)out_size; (void)ws_size;

  const float* x   = (const float*)d_in[0];   // [16,512,64,64]
  const float* cen = (const float*)d_in[1];   // [512,192]
  const float* w1  = (const float*)d_in[2];   // [512,512]
  const float* b1  = (const float*)d_in[3];
  const float* w2  = (const float*)d_in[4];
  const float* g2  = (const float*)d_in[5];
  const float* be2 = (const float*)d_in[6];
  const float* w3  = (const float*)d_in[7];
  const float* b3  = (const float*)d_in[8];
  const float* g3  = (const float*)d_in[9];
  const float* be3 = (const float*)d_in[10];
  float* out = (float*)d_out;

  const long long S1 = 16LL * 512 * 4096;      // [B,C,N]
  const long long S2 = 16LL * 192 * 4096;      // attT [B,M,N]

  float* ws   = (float*)d_ws;
  float* x1   = ws;                            // conv1 out; reused for y3
  float* attT = ws + S1;                       // [16][192][4096]
  float* xre  = ws + S1 + S2;
  float* y2   = ws + 2 * S1 + S2;              // conv2 out; BN+res in place
  float* rs   = ws + 3 * S1 + S2;              // [16*192] row sums
  float* st   = rs + 16 * 192;                 // [2*512] BN stats

  const long long sBX = 512LL * 4096;          // per-batch [C,N] stride
  const long long sAT = 192LL * 4096;          // per-batch attT stride

  const dim3 blk(256);
  const dim3 gConv(4096 / 128, 512 / 64, 16);  // (32, 8, 16)
  const dim3 gAtt (4096 / 128, 192 / 64, 16);  // (32, 3, 16)

  // 1. x1 = W1 @ x + b1                               [512 x 4096] per batch
  hipLaunchKernelGGL((gemm_wmma<0>), gConv, blk, 0, stream,
                     w1, x, b1, x1, 512, 512, 4096, 4096, 0LL, sBX, sBX);
  // 2. attT = centers^T @ x1                          [192 x 4096] per batch
  hipLaunchKernelGGL((gemm_wmma<1>), gAtt, blk, 0, stream,
                     cen, x1, (const float*)nullptr, attT, 512, 192, 4096, 4096,
                     0LL, sBX, sAT);
  // 3. softmax over K (columns of attT)
  hipLaunchKernelGGL(softmax_k, dim3(16 * 4096 / 256), blk, 0, stream, attT);
  // 4-5. double normalization over spatial positions
  hipLaunchKernelGGL(rowsum_kernel, dim3(16 * 192), blk, 0, stream, attT, rs);
  hipLaunchKernelGGL(scale_att, dim3(4096), blk, 0, stream,
                     (float4*)attT, rs, S2 / 4);
  // 6. x_re = centers @ attT                          [512 x 4096] per batch
  hipLaunchKernelGGL((gemm_wmma<0>), gConv, blk, 0, stream,
                     cen, attT, (const float*)nullptr, xre, 192, 192, 4096, 4096,
                     0LL, sAT, sBX);
  // 7. y2 = W2 @ x_re
  hipLaunchKernelGGL((gemm_wmma<0>), gConv, blk, 0, stream,
                     w2, xre, (const float*)nullptr, y2, 512, 512, 4096, 4096,
                     0LL, sBX, sBX);
  // 8-9. BN + ReLU + residual (in place)
  hipLaunchKernelGGL(bn_stats, dim3(512), blk, 0, stream, y2, st);
  hipLaunchKernelGGL(bn_relu_res, dim3(8192), blk, 0, stream,
                     (const float4*)y2, st, g2, be2, (const float4*)x, (float4*)y2,
                     S1 / 4);
  // 10. y3 = W3 @ t + b3   (reuse x1)
  hipLaunchKernelGGL((gemm_wmma<0>), gConv, blk, 0, stream,
                     w3, y2, b3, x1, 512, 512, 4096, 4096, 0LL, sBX, sBX);
  // 11-12. BN + ReLU -> out
  hipLaunchKernelGGL(bn_stats, dim3(512), blk, 0, stream, x1, st);
  hipLaunchKernelGGL(bn_relu_res, dim3(8192), blk, 0, stream,
                     (const float4*)x1, st, g3, be3, (const float4*)nullptr,
                     (float4*)out, S1 / 4);
}
